// Conv1d_65652870087545
// MI455X (gfx1250) — matrix operand order repaired and checked by reading, not compiled, not run on hardware
//
#include <hip/hip_runtime.h>
#include <hip/hip_bf16.h>

// ---- CDNA5 (gfx1250) types ----
typedef __attribute__((ext_vector_type(16))) __bf16        bf16x16;
typedef __attribute__((ext_vector_type(8)))  float         v8f;
typedef __attribute__((ext_vector_type(8)))  unsigned int  u32x8;
typedef __attribute__((ext_vector_type(4)))  unsigned int  u32x4;
typedef __attribute__((ext_vector_type(8)))  int           i32x8;
typedef __attribute__((ext_vector_type(4)))  int           i32x4;

#define B_SZ   32
#define C_IN   8
#define L_SZ   16384
#define C_OUT  64
#define K_SZ   512
#define L_OUT  15873          // L - K + 1
#define KK     4096           // C_IN * K
#define NCHUNK 128            // KK / 32
#define TN     128            // t-columns per workgroup (8 waves x 16)
#define WIN    640            // signal window per channel: TN-1 + 511 + rounding
#define NT_TILES 125          // ceil(L_OUT / TN)
#define W_ELEMS (4*NCHUNK*32*16)   // 262144 packed weight elements per plane

// ------------------------------------------------------------------
// Prepass: repack weight f32 -> (bf16_hi, bf16_lo) planes laid out in
// the exact per-lane order of the WMMA A-fragment, so the compute
// kernel loads each fragment with two aligned b128 loads.
// A-frag (16-bit A 16x32): lane(r=lane&15->row o, h=lane>>4),
//   element j  <->  K_local = 16*(j>>3) + 8*h + (j&7)
// ------------------------------------------------------------------
__global__ void pack_weights(const float* __restrict__ w,
                             __bf16* __restrict__ whi,
                             __bf16* __restrict__ wlo) {
    int tid  = blockIdx.x * 256 + threadIdx.x;     // 262144 threads
    int j    = tid & 15;
    int lane = (tid >> 4) & 31;
    int c    = (tid >> 9) & 127;                   // K-chunk
    int mt   = tid >> 16;                          // o-tile 0..3
    int hl   = lane >> 4;
    int r    = lane & 15;
    int kk   = c * 32 + ((j >> 3) << 4) + hl * 8 + (j & 7);
    int o    = mt * 16 + r;
    float x  = w[(size_t)(o * C_IN + (kk >> 9)) * K_SZ + (kk & 511)];
    __bf16 h = (__bf16)x;                          // round-to-nearest
    whi[tid] = h;
    wlo[tid] = (__bf16)(x - (float)h);             // residual
}

// ------------------------------------------------------------------
// Main kernel: implicit-GEMM valid correlation via v_wmma_f32_16x16x32_bf16
// with hi/lo split (3 products) for ~fp32 accuracy.
// Workgroup: 8 waves; wave wv computes t = t0 + wv*16 + [0,16) for all 64 o.
// Signal window staged by TDM (tensor_load_to_lds), one channel per wave.
// ------------------------------------------------------------------
__global__ __launch_bounds__(256) void conv_wmma(const float* __restrict__ sig,
                                                 const float* __restrict__ bias,
                                                 const __bf16* __restrict__ whi,
                                                 const __bf16* __restrict__ wlo,
                                                 float* __restrict__ out) {
    __shared__ float win[C_IN][WIN];               // 20.5 KB raw f32 window

    const int b    = blockIdx.y;
    const int t0   = blockIdx.x * TN;
    const int lane = threadIdx.x & 31;
    const int wv   = threadIdx.x >> 5;             // wave id == channel id
    const int hl   = lane >> 4;
    const int n    = lane & 15;

    // ---- TDM stage: wave wv DMAs sig[b, wv, t0 .. t0+WIN) into LDS.
    // tensor_dim0 = remaining elements so OOB tail reads return zero.
    {
        unsigned long long ga =
            (unsigned long long)(const void*)(sig + (size_t)(b * C_IN + wv) * L_SZ + t0);
        unsigned int lds_off = (unsigned int)(unsigned long long)(&win[wv][0]);
        unsigned int remain  = (unsigned int)(L_SZ - t0);

        u32x4 g0; i32x8 g1;
        i32x4 g2 = (i32x4)0; i32x4 g3 = (i32x4)0; i32x8 g4 = (i32x8)0;
        g0.x = 1u;                                          // count=1, user desc
        g0.y = lds_off;                                     // lds_addr (bytes)
        g0.z = (unsigned int)ga;                            // global_addr[31:0]
        g0.w = (unsigned int)(ga >> 32) | (2u << 30);       // addr[56:32] | type=2
        g1.s0 = (int)0x00020000u;                           // data_size=2 (4B), no mask
        g1.s1 = (int)(remain << 16);                        // tensor_dim0[15:0]
        g1.s2 = (int)((remain >> 16) | (1u << 16));         // dim0[31:16] | tensor_dim1=1
        g1.s3 = (int)((unsigned)WIN << 16);                 // tile_dim0 = WIN
        g1.s4 = 1;                                          // tile_dim1=1, tile_dim2=0
        g1.s5 = L_SZ;                                       // tensor_dim0_stride
        g1.s6 = 0;
        g1.s7 = 0;
        __builtin_amdgcn_tensor_load_to_lds(g0, g1, g2, g3, g4, 0);
        __builtin_amdgcn_s_wait_tensorcnt(0);
    }
    __syncthreads();

    v8f acc[4] = {};                                        // 4 o-tiles x 16 t, f32
    const u32x8* __restrict__ wh = (const u32x8*)whi;       // 32B A-fragments
    const u32x8* __restrict__ wl = (const u32x8*)wlo;
    const int tloc = wv * 16 + n;                           // t offset in tile

    for (int c = 0; c < NCHUNK; ++c) {
        const int ch  = c >> 4;                             // input channel
        const int pos = tloc + ((c & 15) << 5) + (hl << 3); // window index (dword aligned)

        // Build B fragment (32x16 sig tile) hi/lo from raw f32 in LDS.
        bf16x16 bhi, blo;
        #pragma unroll
        for (int j = 0; j < 16; ++j) {
            float x = win[ch][pos + (j & 7) + ((j >> 3) << 4)];
            __bf16 h = (__bf16)x;
            bhi[j] = h;
            blo[j] = (__bf16)(x - (float)h);
        }

        // 4 o-tiles x 3 split products
        #pragma unroll
        for (int mt = 0; mt < 4; ++mt) {
            const int fidx = (mt * NCHUNK + c) * 32 + lane;
            bf16x16 ah = __builtin_bit_cast(bf16x16, wh[fidx]);
            bf16x16 al = __builtin_bit_cast(bf16x16, wl[fidx]);
            acc[mt] = __builtin_amdgcn_wmma_f32_16x16x32_bf16(
                          false, ah, false, bhi, (short)0, acc[mt], false, false);
            acc[mt] = __builtin_amdgcn_wmma_f32_16x16x32_bf16(
                          false, ah, false, blo, (short)0, acc[mt], false, false);
            acc[mt] = __builtin_amdgcn_wmma_f32_16x16x32_bf16(
                          false, al, false, bhi, (short)0, acc[mt], false, false);
        }
    }

    // D layout: VGPR r -> row M = r + 8*hl, col N = n. Add bias, store.
    const int t = t0 + tloc;
    if (t < L_OUT) {
        #pragma unroll
        for (int mt = 0; mt < 4; ++mt) {
            #pragma unroll
            for (int r = 0; r < 8; ++r) {
                const int o = mt * 16 + hl * 8 + r;
                out[(size_t)(b * C_OUT + o) * L_OUT + t] = acc[mt][r] + bias[o];
            }
        }
    }
}

extern "C" void kernel_launch(void* const* d_in, const int* in_sizes, int n_in,
                              void* d_out, int out_size, void* d_ws, size_t ws_size,
                              hipStream_t stream) {
    const float* sig  = (const float*)d_in[0];   // (32, 8, 16384)
    const float* w    = (const float*)d_in[1];   // (64, 8, 512)
    const float* bias = (const float*)d_in[2];   // (1, 64, 1)
    float* out        = (float*)d_out;           // (32, 64, 15873)

    __bf16* whi = (__bf16*)d_ws;                 // 512 KB
    __bf16* wlo = whi + W_ELEMS;                 // 512 KB

    pack_weights<<<W_ELEMS / 256, 256, 0, stream>>>(w, whi, wlo);

    dim3 grid(NT_TILES, B_SZ);
    conv_wmma<<<grid, 256, 0, stream>>>(sig, bias, whi, wlo, out);
}
